// Attention_17042430230961
// MI455X (gfx1250) — matrix operand order repaired and checked by reading, not compiled, not run on hardware
//
#include <hip/hip_runtime.h>
#include <hip/hip_bf16.h>
#include <cstdint>

// ---------------------------------------------------------------------------
// Problem constants (from reference): B=4, N=4096, C=128, H=4, D=32
// ---------------------------------------------------------------------------
#define BATCH   4
#define NSEQ    4096
#define CDIM    128
#define HEADS   4
#define DHEAD   32
#define NROWS   (BATCH * NSEQ)       // 16384 tokens
#define QKVC    (3 * CDIM)           // 384

typedef __attribute__((ext_vector_type(16))) _Float16 v16h;
typedef __attribute__((ext_vector_type(8)))  float    v8f;

union AFrag { v16h v; uint4 q[2]; };           // 16x32 f16 A/B operand (8 VGPRs)
union CFrag { v8f  v; float f[8]; };           // 16x16 f32 C/D operand (8 VGPRs)
union H8    { uint4 q; _Float16 h[8]; };

// CDNA5 async global->LDS copy (ASYNCcnt path), guarded so we fall back to a
// plain synchronous copy if this toolchain does not expose the builtin.
// Signature (verified by compile probe): (AS1 v4i* gsrc, AS3 v4i* ldst,
// imm offset, imm cpol) where v4i is the GCC-vector int4 type.
#if defined(__has_builtin)
#  if __has_builtin(__builtin_amdgcn_global_load_async_to_lds_b128)
#    define HAVE_ASYNC_LDS 1
#  endif
#endif
#ifndef HAVE_ASYNC_LDS
#  define HAVE_ASYNC_LDS 0
#endif

typedef int v4i __attribute__((vector_size(16)));
#define AS_GLOBAL __attribute__((address_space(1)))
#define AS_LDS    __attribute__((address_space(3)))

// 16-lane (half-wave) reductions — matches C-fragment layout where one output
// row lives in one VGPR across lanes 0-15 (or 16-31).
__device__ __forceinline__ float redmax16(float v) {
  v = fmaxf(v, __shfl_xor(v, 1, 32));
  v = fmaxf(v, __shfl_xor(v, 2, 32));
  v = fmaxf(v, __shfl_xor(v, 4, 32));
  v = fmaxf(v, __shfl_xor(v, 8, 32));
  return v;
}
__device__ __forceinline__ float redsum16(float v) {
  v += __shfl_xor(v, 1, 32);
  v += __shfl_xor(v, 2, 32);
  v += __shfl_xor(v, 4, 32);
  v += __shfl_xor(v, 8, 32);
  return v;
}

// ---------------------------------------------------------------------------
// Scratch zeroing (harness poisons workspace with 0xAA)
// ---------------------------------------------------------------------------
__global__ void zero_f32(float* __restrict__ p, int n) {
  int i = blockIdx.x * blockDim.x + threadIdx.x;
  if (i < n) p[i] = 0.0f;
}

// ---------------------------------------------------------------------------
// WMMA GEMM: out[M,NCOLS] = A[M,128] * W[128,NCOLS] (+bias)
// Block = 128 threads (4 waves), tile 64(M) x 64(N), K staged in LDS as f16.
// NCOLS / HAS_BIAS / HAS_SUMSQ are compile-time so the epilogue is branch-free
// and store addresses strength-reduce to base + immediate offsets.
// ---------------------------------------------------------------------------
template <int NCOLS, bool HAS_BIAS, bool HAS_SUMSQ>
__global__ __launch_bounds__(128) void gemm_wmma(
    const float* __restrict__ A, const float* __restrict__ W,
    float* __restrict__ out, const float* __restrict__ bias,
    float* __restrict__ sumsq_q, float* __restrict__ sumsq_k)
{
  __shared__ __align__(16) _Float16 At[64 * 32];   // 64 rows x 32 K, row-major
  __shared__ __align__(16) _Float16 Bt[64 * 32];   // 64 cols x 32 K (transposed)
  __shared__ float colsq[64];

  const int t    = threadIdx.x;
  const int w    = t >> 5;
  const int lane = t & 31;
  const int ln   = lane & 15;          // position within half-wave
  const int hs   = lane >> 4;          // which half of the wave
  const int rowBase = blockIdx.x * 64;
  const int colBase = blockIdx.y * 64;

  CFrag acc[4] = {};                   // 16 x 64 output strip per wave

  #pragma unroll
  for (int k0 = 0; k0 < 128; k0 += 32) {
    __syncthreads();
    { // stage A tile (f32 -> f16): 16 contiguous elements per thread
      int ar = t >> 1, cs = (t & 1) * 16;
      const float* src = A + (size_t)(rowBase + ar) * 128 + k0 + cs;
      #pragma unroll
      for (int i = 0; i < 16; ++i) At[ar * 32 + cs + i] = (_Float16)src[i];
    }
    { // stage B tile: COALESCED row-major reads of W, transpose on LDS store
      //   Bt[col][k] = W[k0+k][colBase+col]
      int kk = t >> 2;                 // K row within chunk (0..31)
      int nc = (t & 3) * 16;           // column group start
      const float* src = W + (size_t)(k0 + kk) * NCOLS + colBase + nc;
      #pragma unroll
      for (int i = 0; i < 16; ++i) Bt[(nc + i) * 32 + kk] = (_Float16)src[i];
    }
    __syncthreads();

    // A fragment per ISA 16-bit 16x32 layout: lanes0-15 K=8h.., K=16+8h..
    AFrag af;
    const uint4* pa = (const uint4*)(At + (w * 16 + ln) * 32);
    af.q[0] = pa[hs];
    af.q[1] = pa[2 + hs];

    #pragma unroll
    for (int c = 0; c < 4; ++c) {
      // B fragment: lane holds column n, K = 8*hs.., 16+8*hs.. (matches A)
      AFrag bf;
      const uint4* pb = (const uint4*)(Bt + (c * 16 + ln) * 32);
      bf.q[0] = pb[hs];
      bf.q[1] = pb[2 + hs];
      acc[c].v = __builtin_amdgcn_wmma_f32_16x16x32_f16(
          false, af.v, false, bf.v, (short)0, acc[c].v, false, false);
    }
  }

  // Epilogue: C layout row = r + 8*hs, col = c*16 + ln. Bias hoisted per column.
  #pragma unroll
  for (int c = 0; c < 4; ++c) {
    const int col = colBase + c * 16 + ln;
    float bv = 0.0f;
    if constexpr (HAS_BIAS) bv = bias[col];
    float* dst = out + (size_t)(rowBase + w * 16 + 8 * hs) * NCOLS + col;
    #pragma unroll
    for (int r = 0; r < 8; ++r)
      dst[(size_t)r * NCOLS] = acc[c].f[r] + bv;
  }

  // Column sum-of-squares for the sequence-axis l2norm (q cols 0-127, k 128-255)
  if constexpr (HAS_SUMSQ) {
    if (colBase < 256) {
      __syncthreads();
      if (t < 64) colsq[t] = 0.0f;
      __syncthreads();
      #pragma unroll
      for (int c = 0; c < 4; ++c) {
        float ss = 0.0f;
        #pragma unroll
        for (int r = 0; r < 8; ++r) { float v = acc[c].f[r]; ss += v * v; }
        atomicAdd(&colsq[c * 16 + ln], ss);        // ds_add_f32, 4 per lane
      }
      __syncthreads();
      if (t < 64) {
        int b = rowBase >> 12;                     // 4096 rows per batch
        float* dst = (colBase < 128)
                   ? (sumsq_q + b * 128 + colBase + t)
                   : (sumsq_k + b * 128 + (colBase - 128) + t);
        atomicAdd(dst, colsq[t]);                  // global_atomic_add_f32
      }
    }
  }
}

// ---------------------------------------------------------------------------
// Normalize q,k over the sequence axis, split qkv, convert to f16 [B,H,N,32].
// The attention SCALE (10.0) is folded into Q here: S = (10*Qn) Kn^T.
// ---------------------------------------------------------------------------
__global__ __launch_bounds__(256) void qkv_norm(
    const float* __restrict__ qkv,
    const float* __restrict__ ssq, const float* __restrict__ ssk,
    _Float16* __restrict__ Qn, _Float16* __restrict__ Kn, _Float16* __restrict__ Vn)
{
  int idx = blockIdx.x * 256 + threadIdx.x;       // exactly NROWS*384 threads
  int row = idx / 384, e = idx % 384;
  int b = row >> 12, n = row & 4095;
  float v = qkv[idx];
  if (e < 128) {
    float dn = fmaxf(sqrtf(ssq[b * 128 + e]), 1e-12f);
    int h = e >> 5, d = e & 31;
    Qn[((size_t)(b * HEADS + h) * NSEQ + n) * DHEAD + d] = (_Float16)((v / dn) * 10.0f);
  } else if (e < 256) {
    int e2 = e - 128;
    float dn = fmaxf(sqrtf(ssk[b * 128 + e2]), 1e-12f);
    int h = e2 >> 5, d = e2 & 31;
    Kn[((size_t)(b * HEADS + h) * NSEQ + n) * DHEAD + d] = (_Float16)(v / dn);
  } else {
    int e2 = e - 256;
    int h = e2 >> 5, d = e2 & 31;
    Vn[((size_t)(b * HEADS + h) * NSEQ + n) * DHEAD + d] = (_Float16)v;
  }
}

// ---------------------------------------------------------------------------
// Flash attention per (b,h): 256 threads = 8 waves, 128 query rows per block,
// 64-key tiles. Q strip lives in registers as an A-fragment for the whole
// kernel (K-dim == DHEAD == 32 exactly matches one WMMA).
// ---------------------------------------------------------------------------
__global__ __launch_bounds__(256) void flash_attn(
    const _Float16* __restrict__ Qn, const _Float16* __restrict__ Kn,
    const _Float16* __restrict__ Vn, float* __restrict__ O)
{
  __shared__ __align__(16) _Float16 Kt[64 * 32];       // 64 keys x 32 dims
  __shared__ __align__(16) _Float16 Vt[32 * 64];       // transposed: dim x key
  __shared__ __align__(16) _Float16 Pt[8][16 * 64];    // per-wave P tile

  const int t    = threadIdx.x;
  const int w    = t >> 5;
  const int lane = t & 31;
  const int ln   = lane & 15;
  const int hs   = lane >> 4;
  const int b    = blockIdx.z, h = blockIdx.y;
  const int qBase = blockIdx.x * 128;
  const size_t headOff = (size_t)(b * HEADS + h) * NSEQ;

  // Q strip (16 rows x 32 dims) as A fragment: row length 64B -> aligned b128s
  AFrag qf;
  {
    const uint4* p = (const uint4*)(Qn + (headOff + qBase + w * 16 + ln) * DHEAD);
    qf.q[0] = p[hs];
    qf.q[1] = p[2 + hs];
  }

  float mrow[8], lrow[8];
  CFrag o0 = {}, o1 = {};
  #pragma unroll
  for (int r = 0; r < 8; ++r) { mrow[r] = -1e30f; lrow[r] = 0.0f; }

  _Float16* Pw = Pt[w];

  for (int j = 0; j < NSEQ; j += 64) {
    __syncthreads();
    // K tile is 4KB contiguous in [B,H,N,32] layout: one b128 per thread.
#if HAVE_ASYNC_LDS
    __builtin_amdgcn_global_load_async_to_lds_b128(
        (AS_GLOBAL v4i*)(Kn + (headOff + j) * DHEAD + (size_t)t * 8),
        (AS_LDS v4i*)(Kt + t * 8),
        0, 0);
#else
    ((uint4*)Kt)[t] = ((const uint4*)(Kn + (headOff + j) * DHEAD))[t];
#endif
    { // V tile, transposed into Vt[dim][key]
      int kl = t >> 2, dc = (t & 3) * 8;
      H8 u;
      u.q = ((const uint4*)(Vn + (headOff + j + kl) * DHEAD))[t & 3];
      #pragma unroll
      for (int i = 0; i < 8; ++i) Vt[(dc + i) * 64 + kl] = u.h[i];
    }
    // Prefetch next K/V tiles into cache while we compute
    {
      int jn = (j + 64 < NSEQ) ? (j + 64) : j;
      __builtin_prefetch(Kn + (headOff + jn) * DHEAD, 0, 1);  // global_prefetch_b8
      __builtin_prefetch(Vn + (headOff + jn) * DHEAD, 0, 1);
    }
#if HAVE_ASYNC_LDS
    asm volatile("s_wait_asynccnt 0x0" ::: "memory");
#endif
    __syncthreads();

    // S = (10*Q) Kt^T : four 16x16 f32 fragments (K-dim = 32 = DHEAD)
    CFrag s[4];
    #pragma unroll
    for (int f = 0; f < 4; ++f) {
      AFrag bf;  // B operand: lane = key column, K(=dim) 8*hs.., 16+8*hs.. (matches Q)
      const uint4* pb = (const uint4*)(Kt + (f * 16 + ln) * 32);
      bf.q[0] = pb[hs];
      bf.q[1] = pb[2 + hs];
      v8f z = {};
      s[f].v = __builtin_amdgcn_wmma_f32_16x16x32_f16(
          false, qf.v, false, bf.v, (short)0, z, false, false);
    }

    // Online softmax (SCALE already folded into Q)
    #pragma unroll
    for (int r = 0; r < 8; ++r) {
      float mt = fmaxf(fmaxf(s[0].f[r], s[1].f[r]), fmaxf(s[2].f[r], s[3].f[r]));
      mt = redmax16(mt);
      float mn = fmaxf(mrow[r], mt);
      float sc = __expf(mrow[r] - mn);
      mrow[r] = mn;
      float p0 = __expf(s[0].f[r] - mn); s[0].f[r] = p0;
      float p1 = __expf(s[1].f[r] - mn); s[1].f[r] = p1;
      float p2 = __expf(s[2].f[r] - mn); s[2].f[r] = p2;
      float p3 = __expf(s[3].f[r] - mn); s[3].f[r] = p3;
      float rsum = redsum16(p0 + p1 + p2 + p3);
      lrow[r] = lrow[r] * sc + rsum;
      o0.f[r] *= sc;
      o1.f[r] *= sc;
    }

    // Spill P (16 rows x 64 keys) to this wave's LDS tile as f16
    #pragma unroll
    for (int f = 0; f < 4; ++f)
      #pragma unroll
      for (int r = 0; r < 8; ++r)
        Pw[(r + 8 * hs) * 64 + f * 16 + ln] = (_Float16)s[f].f[r];

    __syncthreads();   // cross-lane LDS visibility (uniform, conservative)

    // O += P * V : two K-chunks of 32 keys, two 16-dim output tiles
    #pragma unroll
    for (int c = 0; c < 2; ++c) {
      AFrag pf;  // P as A fragment
      const uint4* pp = (const uint4*)(Pw + ln * 64);
      pf.q[0] = pp[4 * c + 2 * hs];
      pf.q[1] = pp[4 * c + 2 * hs + 1];
      AFrag vf0, vf1;  // V^T rows give contiguous key runs per output dim
      const uint4* pv0 = (const uint4*)(Vt + ln * 64);
      vf0.q[0] = pv0[4 * c + 2 * hs];
      vf0.q[1] = pv0[4 * c + 2 * hs + 1];
      const uint4* pv1 = (const uint4*)(Vt + (16 + ln) * 64);
      vf1.q[0] = pv1[4 * c + 2 * hs];
      vf1.q[1] = pv1[4 * c + 2 * hs + 1];
      o0.v = __builtin_amdgcn_wmma_f32_16x16x32_f16(
          false, pf.v, false, vf0.v, (short)0, o0.v, false, false);
      o1.v = __builtin_amdgcn_wmma_f32_16x16x32_f16(
          false, pf.v, false, vf1.v, (short)0, o1.v, false, false);
    }
  }

  // Final: divide by softmax denominator, write [B,N, h*32+d] f32
  #pragma unroll
  for (int r = 0; r < 8; ++r) {
    int row = qBase + w * 16 + r + 8 * hs;
    float inv = 1.0f / lrow[r];
    size_t base = ((size_t)b * NSEQ + row) * CDIM + h * DHEAD;
    O[base + ln]      = o0.f[r] * inv;
    O[base + 16 + ln] = o1.f[r] * inv;
  }
}

// ---------------------------------------------------------------------------
// Launch
// ---------------------------------------------------------------------------
extern "C" void kernel_launch(void* const* d_in, const int* in_sizes, int n_in,
                              void* d_out, int out_size, void* d_ws, size_t ws_size,
                              hipStream_t stream)
{
  (void)in_sizes; (void)n_in; (void)out_size; (void)ws_size;
  const float* x    = (const float*)d_in[0];   // [4,4096,128]
  const float* Wqkv = (const float*)d_in[1];   // [128,384]
  const float* Wout = (const float*)d_in[2];   // [128,128]
  const float* bout = (const float*)d_in[3];   // [128]
  float* out = (float*)d_out;                  // [4,4096,128]

  char* ws = (char*)d_ws;
  size_t off = 0;
  auto alloc = [&](size_t bytes) -> void* {
    void* p = ws + off;
    off = (off + bytes + 255) & ~(size_t)255;
    return p;
  };
  float*    qkv  = (float*)   alloc((size_t)NROWS * QKVC * sizeof(float)); // 25.2 MB
  float*    ssq  = (float*)   alloc(BATCH * CDIM * sizeof(float));         // contiguous
  float*    ssk  = (float*)   alloc(BATCH * CDIM * sizeof(float));         // with ssq
  _Float16* Qn   = (_Float16*)alloc((size_t)NROWS * CDIM * sizeof(_Float16));
  _Float16* Kn   = (_Float16*)alloc((size_t)NROWS * CDIM * sizeof(_Float16));
  _Float16* Vn   = (_Float16*)alloc((size_t)NROWS * CDIM * sizeof(_Float16));
  float*    Obuf = (float*)   alloc((size_t)NROWS * CDIM * sizeof(float)); // 8.4 MB

  // 1) zero the column sum-of-squares accumulators (ssq and ssk are adjacent)
  zero_f32<<<4, 256, 0, stream>>>(ssq, 2 * BATCH * CDIM);

  // 2) QKV projection + column sumsq (WMMA f16)
  gemm_wmma<QKVC, false, true><<<dim3(NROWS / 64, QKVC / 64), 128, 0, stream>>>(
      x, Wqkv, qkv, nullptr, ssq, ssk);

  // 3) sequence-axis l2norm + split + f16 convert (SCALE folded into Q)
  qkv_norm<<<(NROWS * QKVC) / 256, 256, 0, stream>>>(qkv, ssq, ssk, Qn, Kn, Vn);

  // 4) flash attention (WMMA f16), one block per 128 query rows per (b,h)
  flash_attn<<<dim3(NSEQ / 128, HEADS, BATCH), 256, 0, stream>>>(Qn, Kn, Vn, Obuf);

  // 5) output projection + bias (WMMA f16)
  gemm_wmma<CDIM, true, false><<<dim3(NROWS / 64, CDIM / 64), 128, 0, stream>>>(
      Obuf, Wout, out, bout, nullptr, nullptr);
}
